// SimOTAAssigner_19645180412370
// MI455X (gfx1250) — matrix operand order repaired and checked by reading, not compiled, not run on hardware
//
#include <hip/hip_runtime.h>
#include <hip/hip_bf16.h>

// ---------------- problem constants ----------------
#define BS   16
#define NA   33600
#define NG   64
#define NC   80
#define KPAD 96          // classes padded to 3x32 for WMMA K-chunks
#define BA   320         // anchors per block (33600 = 320 * 105, no tail)
#define NBLK 105
#define NTH  320         // 10 waves of 32
#define TOPK 10
#define FLTMAX 3.402823466e+38f

typedef __attribute__((ext_vector_type(16))) _Float16 v16h;
typedef __attribute__((ext_vector_type(8)))  float    v8f;

// ---- CDNA5 async global->LDS copy (ASYNCcnt-tracked), via inline asm ----
// GLOBAL_LOAD_ASYNC_TO_LDS_B128: VDST = per-lane LDS byte address, VADDR =
// 64-bit global address, SADDR = off. Flat->LDS maps by low-32-bit truncation.
__device__ __forceinline__ void async_copy_b128(const float* gsrc, float* ldst) {
    unsigned lds_addr = (unsigned)(unsigned long long)ldst;
    asm volatile("global_load_async_to_lds_b128 %0, %1, off"
                 :: "v"(lds_addr), "v"(gsrc)
                 : "memory");
}
__device__ __forceinline__ void async_wait0() {
#if __has_builtin(__builtin_amdgcn_s_wait_asynccnt)
    __builtin_amdgcn_s_wait_asynccnt(0);
#else
    asm volatile("s_wait_asynccnt 0" ::: "memory");
#endif
}

// Dynamic-LDS layout for kernels 1 and 3 (~234 KB; CDNA5 WGP has 320 KB LDS).
// cost+iou (contiguous, 163 KB) double as the raw-score async staging buffer
// (BA*NC*4 = 100 KB) before they are produced — usage is barrier-separated.
struct Smem {
    _Float16 diff[BA][KPAD];   // f16 (log1mp - logp) per anchor x class
    float    cost[NG][BA];     // staging -> WMMA gather -> cost (in place)
    float    iou [NG][BA];     // staging -> masked ious
    float    basev[BA];        // -sum_c log1mp
    float    gtb[NG][4];
    float    gcx[NG], gcy[NG], gat[NG];
    int      glab[NG];
    int      gvalid[NG];
    unsigned mlo[BA], mhi[BA]; // 64-bit match mask per local anchor
    int      sel_lab[BA], sel_mg[BA], sel_fg[BA];
    float    sel_pi[BA];
    int      ldynk[NG];
    int      lsel[NG][TOPK];
};

// ---- shared compute: fill s.cost / s.iou tiles for (batch b, anchors [abase, abase+320)) ----
__device__ __forceinline__ void compute_tiles(
    Smem& s,
    const float* __restrict__ pd_scores, const float* __restrict__ pd_bboxes,
    const float* __restrict__ anc, const int* __restrict__ gt_labels,
    const float* __restrict__ gt_bboxes, const int* __restrict__ mask_gt,
    const float* __restrict__ stride, int b, int abase)
{
    const int tid  = threadIdx.x;
    const int lane = tid & 31;
    const int wave = tid >> 5;

    float* rawsc = &s.cost[0][0];   // BA*NC floats staging (overlaps cost+iou)
    const float* sc_base = pd_scores + ((size_t)b * NA + abase) * NC;

    // ---- stage the 320x80 f32 score tile into LDS (async, coalesced) ----
    for (int e = tid; e < (BA * NC) / 4; e += NTH) {
        async_copy_b128(sc_base + (size_t)e * 4, rawsc + (size_t)e * 4);
    }
    __builtin_prefetch(sc_base + (size_t)BA * NC, 0, 1);  // next block's tile

    // GT data -> LDS (while async copy is in flight)
    if (tid < NG) {
        const int g = tid;
        const float* gb = gt_bboxes + ((size_t)b * NG + g) * 4;
        float x1 = gb[0], y1 = gb[1], x2 = gb[2], y2 = gb[3];
        s.gtb[g][0] = x1; s.gtb[g][1] = y1; s.gtb[g][2] = x2; s.gtb[g][3] = y2;
        s.gcx[g] = 0.5f * (x1 + x2);
        s.gcy[g] = 0.5f * (y1 + y2);
        s.gat[g] = atanf((x2 - x1) / ((y2 - y1) + 1e-7f));
        s.glab[g]   = gt_labels[(size_t)b * NG + g];
        s.gvalid[g] = (mask_gt[(size_t)b * NG + g] != 0) ? 1 : 0;
    }

    async_wait0();
    __syncthreads();

    // ---- scores -> f16 diff + base ----
    {
        float bsum = 0.f;
        for (int c = 0; c < NC; ++c) {
            float p   = sqrtf(rawsc[tid * NC + c]);
            float lp  = fmaxf(logf(p), -100.f);
            float l1m = fmaxf(log1pf(-p), -100.f);
            bsum -= l1m;
            s.diff[tid][c] = (_Float16)(l1m - lp);
        }
        for (int c = NC; c < KPAD; ++c) s.diff[tid][c] = (_Float16)0.f;
        s.basev[tid] = bsum;
    }
    __syncthreads();

    // ---- WMMA: gather[g][a] = sum_c onehot(lab_g, c) * diff[a][c] ----
    // B (one-hot) hoisted per gt tile; each wave covers subtiles {wave, wave+10}.
    {
        const int half = (lane >= 16) ? 1 : 0;
        const int ln   = lane & 15;
        #pragma unroll
        for (int gtt = 0; gtt < 4; ++gtt) {
            const int lab = s.glab[gtt * 16 + ln];
            v16h B0, B1, B2;
            #pragma unroll
            for (int j = 0; j < 16; ++j) {
                // B k order used here (contiguous, same as A): lanes<16 K=0..15, lanes>=16 K=16..31
                int kb = j + (half ? 16 : 0);
                B0[j] = (_Float16)((kb      == lab) ? 1.0f : 0.0f);
                B1[j] = (_Float16)((kb + 32 == lab) ? 1.0f : 0.0f);
                B2[j] = (_Float16)((kb + 64 == lab) ? 1.0f : 0.0f);
            }
            for (int st = wave; st < 20; st += 10) {
                const int arow = st * 16 + ln;
                v8f acc = {};
                #pragma unroll
                for (int kc = 0; kc < 3; ++kc) {
                    v16h A;
                    #pragma unroll
                    for (int j = 0; j < 16; ++j) {
                        // A k order used here (contiguous, same as B): lanes<16 K=0..15, lanes>=16 K=16..31
                        int ka = kc * 32 + j + (half ? 16 : 0);
                        A[j] = s.diff[arow][ka];
                    }
                    acc = __builtin_amdgcn_wmma_f32_16x16x32_f16(
                        false, A, false, (kc == 0) ? B0 : (kc == 1) ? B1 : B2,
                        (short)0, acc, false, false);
                }
                #pragma unroll
                for (int r = 0; r < 8; ++r) {
                    int m = st * 16 + r + (half ? 8 : 0);  // anchor (local)
                    int n = gtt * 16 + ln;                 // gt
                    s.cost[n][m] = acc[r];
                }
            }
        }
    }
    __syncthreads();

    // ---- geometry + final cost/iou per (gt, anchor) ----
    {
        const int a = abase + tid;
        const float* pb = pd_bboxes + ((size_t)b * NA + a) * 4;
        const float px1 = pb[0], py1 = pb[1], px2 = pb[2], py2 = pb[3];
        const float w2 = px2 - px1, h2 = py2 - py1;
        const float pat = atanf(w2 / (h2 + 1e-7f));
        const float ax = anc[(size_t)a * 2 + 0], ay = anc[(size_t)a * 2 + 1];
        const float cd = stride[(size_t)b * NA + a] * 2.5f;

        unsigned inlo = 0u, inhi = 0u;
        for (int g = 0; g < NG; ++g) {
            bool ic = (fabsf(ax - s.gcx[g]) < cd) && (fabsf(ay - s.gcy[g]) < cd)
                      && (s.gvalid[g] != 0);
            if (ic) { if (g < 32) inlo |= 1u << g; else inhi |= 1u << (g - 32); }
        }
        const bool af = (inlo | inhi) != 0u;   // anchor_filter
        const float base = s.basev[tid];

        for (int g = 0; g < NG; ++g) {
            float gx1 = s.gtb[g][0], gy1 = s.gtb[g][1];
            float gx2 = s.gtb[g][2], gy2 = s.gtb[g][3];
            float iw = fmaxf(fminf(gx2, px2) - fmaxf(gx1, px1), 0.f);
            float ih = fmaxf(fminf(gy2, py2) - fmaxf(gy1, py1), 0.f);
            float inter = iw * ih;
            float w1 = gx2 - gx1, h1 = gy2 - gy1;
            float uni = w1 * h1 + w2 * h2 - inter + 1e-7f;
            float iouv = inter / uni;
            float cw = fmaxf(gx2, px2) - fminf(gx1, px1);
            float ch = fmaxf(gy2, py2) - fminf(gy1, py1);
            float c2 = cw * cw + ch * ch + 1e-7f;
            float dx = (px1 + px2 - gx1 - gx2);
            float dy = (py1 + py2 - gy1 - gy2);
            float rho2 = 0.25f * (dx * dx + dy * dy);
            float dv = pat - s.gat[g];
            float v  = 0.40528473456935108577f * dv * dv;  // 4/pi^2 * dv^2
            float alpha = v / (v - iouv + 1.0f + 1e-7f);
            float ciou = iouv - (rho2 / c2 + v * alpha);

            bool  gv  = (s.gvalid[g] != 0);
            float ious = (af && gv) ? fmaxf(ciou, 0.f) : 0.f;
            float il   = -logf(ious + 1e-8f);
            bool  ic   = (g < 32) ? ((inlo >> g) & 1u) : ((inhi >> (g - 32)) & 1u);
            float cost = base + s.cost[g][tid] + 3.0f * il
                       + (ic ? 0.f : 1.0e6f)
                       + (af ? 0.f : 1.0e8f)
                       + (gv ? 0.f : 1.0e9f);
            s.cost[g][tid] = cost;
            s.iou [g][tid] = ious;
        }
    }
    __syncthreads();
}

// ---------------- kernel 1: tiles + per-block per-gt top-10 ----------------
__global__ __launch_bounds__(NTH) void k1_topk(
    const float* __restrict__ pd_scores, const float* __restrict__ pd_bboxes,
    const float* __restrict__ anc, const int* __restrict__ gt_labels,
    const float* __restrict__ gt_bboxes, const int* __restrict__ mask_gt,
    const float* __restrict__ stride,
    float* __restrict__ blk_cost, int* __restrict__ blk_idx,
    float* __restrict__ blk_iou)
{
    extern __shared__ char raw[];
    Smem& s = *reinterpret_cast<Smem*>(raw);
    const int blk = blockIdx.x, b = blockIdx.y;
    const int abase = blk * BA;
    compute_tiles(s, pd_scores, pd_bboxes, anc, gt_labels, gt_bboxes,
                  mask_gt, stride, b, abase);

    const int lane = threadIdx.x & 31, wave = threadIdx.x >> 5;
    for (int g = wave; g < NG; g += 10) {
        float cv[10], iv[10];
        #pragma unroll
        for (int k = 0; k < 10; ++k) {
            cv[k] = s.cost[g][lane + 32 * k];
            iv[k] = s.iou [g][lane + 32 * k];
        }
        const size_t rec = (((size_t)b * NG + g) * NBLK + blk) * TOPK;
        // --- 10 smallest costs (stable by anchor index) ---
        for (int r = 0; r < TOPK; ++r) {
            float bv = FLTMAX; int bp = 0x7FFFFFFF;
            #pragma unroll
            for (int k = 0; k < 10; ++k) {
                int p = lane + 32 * k;
                if (cv[k] < bv) { bv = cv[k]; bp = p; }
            }
            for (int off = 16; off; off >>= 1) {
                float ov = __shfl_xor(bv, off, 32);
                int   op = __shfl_xor(bp, off, 32);
                if (ov < bv || (ov == bv && op < bp)) { bv = ov; bp = op; }
            }
            if ((bp & 31) == lane) {
                int kw = bp >> 5;
                #pragma unroll
                for (int k = 0; k < 10; ++k) if (k == kw) cv[k] = FLTMAX;
            }
            if (lane == 0) { blk_cost[rec + r] = bv; blk_idx[rec + r] = abase + bp; }
        }
        // --- 10 largest ious ---
        for (int r = 0; r < TOPK; ++r) {
            float bv = -FLTMAX; int bp = 0x7FFFFFFF;
            #pragma unroll
            for (int k = 0; k < 10; ++k) {
                int p = lane + 32 * k;
                if (iv[k] > bv) { bv = iv[k]; bp = p; }
            }
            for (int off = 16; off; off >>= 1) {
                float ov = __shfl_xor(bv, off, 32);
                int   op = __shfl_xor(bp, off, 32);
                if (ov > bv || (ov == bv && op < bp)) { bv = ov; bp = op; }
            }
            if ((bp & 31) == lane) {
                int kw = bp >> 5;
                #pragma unroll
                for (int k = 0; k < 10; ++k) if (k == kw) iv[k] = -FLTMAX;
            }
            if (lane == 0) blk_iou[rec + r] = bv;
        }
    }
}

// ---------------- kernel 2: merge block lists -> dyn_k + selected anchors ----------------
__global__ __launch_bounds__(32) void k2_merge(
    const float* __restrict__ blk_cost, const int* __restrict__ blk_idx,
    const float* __restrict__ blk_iou, int* __restrict__ sel)
{
    __shared__ float c [NBLK * TOPK];
    __shared__ int   id[NBLK * TOPK];
    __shared__ float io[NBLK * TOPK];
    const int g = blockIdx.x, b = blockIdx.y, lane = threadIdx.x;
    const size_t rb = ((size_t)b * NG + g) * NBLK * TOPK;
    for (int e = lane; e < NBLK * TOPK; e += 32) {
        c[e] = blk_cost[rb + e]; id[e] = blk_idx[rb + e]; io[e] = blk_iou[rb + e];
    }
    __syncthreads();

    // dyn_k = clamp(floor(sum(top10 iou) + 0.5), 1, 10)
    float isum = 0.f;
    for (int r = 0; r < TOPK; ++r) {
        float bv = -FLTMAX; int bp = 0x7FFFFFFF;
        for (int e = lane; e < NBLK * TOPK; e += 32) {
            float v = io[e]; if (v > bv) { bv = v; bp = e; }
        }
        for (int off = 16; off; off >>= 1) {
            float ov = __shfl_xor(bv, off, 32);
            int   op = __shfl_xor(bp, off, 32);
            if (ov > bv || (ov == bv && op < bp)) { bv = ov; bp = op; }
        }
        if (lane == 0) { io[bp] = -FLTMAX; isum += bv; }
        __syncthreads();
    }
    // top-10 (cost, idx) lexicographic ascending == stable argsort semantics
    const size_t ob = ((size_t)b * NG + g) * (TOPK + 1);
    for (int r = 0; r < TOPK; ++r) {
        float bv = FLTMAX; int bp = 0x7FFFFFFF; int bi = 0x7FFFFFFF;
        for (int e = lane; e < NBLK * TOPK; e += 32) {
            float v = c[e]; int i = id[e];
            if (v < bv || (v == bv && i < bi)) { bv = v; bp = e; bi = i; }
        }
        for (int off = 16; off; off >>= 1) {
            float ov = __shfl_xor(bv, off, 32);
            int   op = __shfl_xor(bp, off, 32);
            int   oi = __shfl_xor(bi, off, 32);
            if (ov < bv || (ov == bv && oi < bi)) { bv = ov; bp = op; bi = oi; }
        }
        if (lane == 0) { c[bp] = FLTMAX; sel[ob + 1 + r] = bi; }
        __syncthreads();
    }
    if (lane == 0) {
        int dynk = (int)(isum + 0.5f);
        if (dynk < 1) dynk = 1; if (dynk > TOPK) dynk = TOPK;
        sel[ob] = dynk;
    }
}

// ---------------- kernel 3: final assignment + outputs ----------------
__global__ __launch_bounds__(NTH) void k3_assign(
    const float* __restrict__ pd_scores, const float* __restrict__ pd_bboxes,
    const float* __restrict__ anc, const int* __restrict__ gt_labels,
    const float* __restrict__ gt_bboxes, const int* __restrict__ mask_gt,
    const float* __restrict__ stride, const int* __restrict__ sel,
    float* __restrict__ out_labels, float* __restrict__ out_boxes,
    float* __restrict__ out_scores, float* __restrict__ out_fg,
    float* __restrict__ out_tgt)
{
    extern __shared__ char raw[];
    Smem& s = *reinterpret_cast<Smem*>(raw);
    const int blk = blockIdx.x, b = blockIdx.y, tid = threadIdx.x;
    const int abase = blk * BA;

    if (tid < NG) {
        const size_t ob = ((size_t)b * NG + tid) * (TOPK + 1);
        s.ldynk[tid] = sel[ob];
        for (int j = 0; j < TOPK; ++j) s.lsel[tid][j] = sel[ob + 1 + j];
    }
    s.mlo[tid] = 0u; s.mhi[tid] = 0u;   // synced inside compute_tiles

    compute_tiles(s, pd_scores, pd_bboxes, anc, gt_labels, gt_bboxes,
                  mask_gt, stride, b, abase);

    // scatter selected anchors into per-anchor match masks
    for (int e = tid; e < NG * TOPK; e += NTH) {
        int g = e / TOPK, j = e % TOPK;
        if (j < s.ldynk[g] && s.gvalid[g]) {
            int loc = s.lsel[g][j] - abase;
            if (loc >= 0 && loc < BA) {
                if (g < 32) atomicOr(&s.mlo[loc], 1u << g);
                else        atomicOr(&s.mhi[loc], 1u << (g - 32));
            }
        }
    }
    __syncthreads();

    {
        unsigned lo = s.mlo[tid], hi = s.mhi[tid];
        int cnt = __popc(lo) + __popc(hi);
        int bg = 0; float bv = s.cost[0][tid];
        for (int g = 1; g < NG; ++g) {
            float v = s.cost[g][tid];
            if (v < bv) { bv = v; bg = g; }
        }
        if (cnt > 1) {  // resolve multi-assignment with cost-argmin one-hot
            lo = (bg < 32) ? (1u << bg) : 0u;
            hi = (bg < 32) ? 0u : (1u << (bg - 32));
        }
        bool fg = (lo | hi) != 0u;
        int  mg = 0;
        if (fg) mg = lo ? (__ffs(lo) - 1) : (32 + __ffs(hi) - 1);
        float pi = 0.f;
        unsigned t = lo;
        while (t) { int g = __ffs(t) - 1;      pi += s.iou[g][tid]; t &= t - 1; }
        t = hi;
        while (t) { int g = __ffs(t) - 1 + 32; pi += s.iou[g][tid]; t &= t - 1; }

        const size_t ai = (size_t)b * NA + abase + tid;
        out_labels[ai] = (float)(fg ? s.glab[mg] : NC);
        out_fg[ai]     = fg ? 1.f : 0.f;
        out_tgt[ai]    = (float)(fg ? mg : 0);
        s.sel_lab[tid] = fg ? s.glab[mg] : -1;
        s.sel_pi [tid] = fg ? pi : 0.f;
        s.sel_mg [tid] = mg;
        s.sel_fg [tid] = fg ? 1 : 0;
    }
    __syncthreads();

    // coalesced cooperative stores
    for (int e = tid; e < BA * 4; e += NTH) {
        int al = e >> 2, d = e & 3;
        float v = s.sel_fg[al] ? s.gtb[s.sel_mg[al]][d] : 0.f;
        out_boxes[((size_t)b * NA + abase) * 4 + e] = v;
    }
    for (int e = tid; e < BA * NC; e += NTH) {
        int al = e / NC, cc = e % NC;
        float v = (cc == s.sel_lab[al]) ? s.sel_pi[al] : 0.f;
        out_scores[((size_t)b * NA + abase) * NC + e] = v;
    }
}

// ---------------- launch ----------------
extern "C" void kernel_launch(void* const* d_in, const int* in_sizes, int n_in,
                              void* d_out, int out_size, void* d_ws, size_t ws_size,
                              hipStream_t stream) {
    const float* pd_scores = (const float*)d_in[0];
    const float* pd_bboxes = (const float*)d_in[1];
    const float* anc       = (const float*)d_in[2];
    const int*   gt_labels = (const int*)  d_in[3];
    const float* gt_bboxes = (const float*)d_in[4];
    const int*   mask_gt   = (const int*)  d_in[5];
    const float* stride    = (const float*)d_in[6];

    const size_t nrec = (size_t)BS * NG * NBLK * TOPK;   // 1,075,200
    float* blk_cost = (float*)d_ws;
    int*   blk_idx  = (int*)  (blk_cost + nrec);
    float* blk_iou  = (float*)(blk_idx  + nrec);
    int*   sel      = (int*)  (blk_iou  + nrec);         // BS*NG*(TOPK+1)

    float* out_labels = (float*)d_out;                       // BS*NA
    float* out_boxes  = out_labels + (size_t)BS * NA;        // BS*NA*4
    float* out_scores = out_boxes  + (size_t)BS * NA * 4;    // BS*NA*NC
    float* out_fg     = out_scores + (size_t)BS * NA * NC;   // BS*NA
    float* out_tgt    = out_fg     + (size_t)BS * NA;        // BS*NA

    const size_t smem = sizeof(Smem);
    k1_topk <<<dim3(NBLK, BS), NTH, smem, stream>>>(
        pd_scores, pd_bboxes, anc, gt_labels, gt_bboxes, mask_gt, stride,
        blk_cost, blk_idx, blk_iou);
    k2_merge <<<dim3(NG, BS), 32, 0, stream>>>(blk_cost, blk_idx, blk_iou, sel);
    k3_assign<<<dim3(NBLK, BS), NTH, smem, stream>>>(
        pd_scores, pd_bboxes, anc, gt_labels, gt_bboxes, mask_gt, stride, sel,
        out_labels, out_boxes, out_scores, out_fg, out_tgt);
}
